// VectorQuantizer_48206712930521
// MI455X (gfx1250) — matrix operand order repaired and checked by reading, not compiled, not run on hardware
//
#include <hip/hip_runtime.h>
#include <stdint.h>

// ---------------------------------------------------------------------------
// Problem constants (shapes fixed by the reference)
// ---------------------------------------------------------------------------
#define NROWS   32768      // 32*1024 flattened input rows
#define DDIM    256        // embedding dim
#define ECODES  8192       // codebook entries
#define NDOUT   (NROWS * DDIM)

// Workspace layout (bytes, all 256-aligned)
#define OFF_FLATN16  0u                      // N*D bf16      = 16,777,216
#define OFF_CODE16   16777216u               // E*D bf16      =  4,194,304
#define OFF_CODE32   20971520u               // E*D f32       =  8,388,608
#define OFF_IDX      29360128u               // N   i32       =    131,072
#define OFF_HIST     29491200u               // E   i32       =     32,768
#define OFF_NORMS    29523968u               // N   f32       =    131,072
#define OFF_SQ       29655040u               // N   f32       =    131,072
#define OFF_SCAL     29786112u               // scalars

typedef __attribute__((ext_vector_type(16))) __bf16 v16bf;
typedef __attribute__((ext_vector_type(8)))  float  v8f;
typedef __attribute__((ext_vector_type(4)))  unsigned int u32x4;
typedef __attribute__((ext_vector_type(8)))  int i32x8;
typedef __attribute__((ext_vector_type(4)))  int i32x4;

struct U32x8 { uint4 lo, hi; };   // 32 bytes, bit-castable to v16bf

// ---- CDNA5 data-movement path selection -----------------------------------
#if __has_builtin(__builtin_amdgcn_tensor_load_to_lds)
  #define HAVE_TDM 1
#elif __has_builtin(__builtin_amdgcn_global_load_async_to_lds_b128)
  #define HAVE_ASYNC 1
#endif

static __device__ __forceinline__ unsigned short f32_to_bf16_rne(float f) {
    unsigned int u = __builtin_bit_cast(unsigned int, f);
    unsigned int lsb = (u >> 16) & 1u;
    u += 0x7fffu + lsb;                 // round-to-nearest-even
    return (unsigned short)(u >> 16);
}

static __device__ __forceinline__ unsigned lds_byte_off(const void* p) {
    // generic LDS pointer = aperture | 32-bit LDS offset
    return (unsigned)(unsigned long long)(uintptr_t)p;
}

#if defined(HAVE_TDM)
// Issue one TDM 2D tile load: `rows` rows of 256 bf16 (512 B), row-major,
// contiguous in both global memory and LDS. Tensor dims == tile dims (no OOB).
// This toolchain's builtin takes 6 args: (g0, g1, g2, g3, g_extra, cpol).
static __device__ __forceinline__ void tdm_load_rows(
    unsigned lds_off, const void* gptr, unsigned rows)
{
    const unsigned long long ga = (unsigned long long)(uintptr_t)gptr;
    u32x4 g0;
    g0.x = 1u;                                   // count=1, user desc, no gather
    g0.y = lds_off;                              // lds_addr (bytes)
    g0.z = (unsigned)ga;                         // global_addr[31:0]
    g0.w = (unsigned)(ga >> 32) | (2u << 30);    // global_addr[56:32] | type=2
    unsigned g1w[8];
    g1w[0] = (1u << 16);                         // workgroup_mask=0, data_size=1 (2B)
    g1w[1] = (256u & 0xffffu) << 16;             // tensor_dim0[15:0]=256
    g1w[2] = ((rows & 0xffffu) << 16);           // tensor_dim0[31:16]=0, tensor_dim1[15:0]=rows
    g1w[3] = (256u << 16);                       // tensor_dim1[31:16]=0, tile_dim0=256
    g1w[4] = (rows & 0xffffu);                   // tile_dim1=rows, tile_dim2=0
    g1w[5] = 256u;                               // tensor_dim0_stride[31:0]=256
    g1w[6] = 0u;                                 // stride0[47:32]=0, stride1[15:0]=0
    g1w[7] = 0u;                                 // stride1[47:16]=0
    const i32x8 g1 = __builtin_bit_cast(i32x8, g1w);
    const i32x4 z4 = {0, 0, 0, 0};               // D# groups 2/3: dims 2-4 unused
    const i32x8 z8 = {0, 0, 0, 0, 0, 0, 0, 0};
    __builtin_amdgcn_tensor_load_to_lds(g0, g1, z4, z4, z8, 0);
}
static __device__ __forceinline__ void wait_tile_loads() {
#if __has_builtin(__builtin_amdgcn_s_wait_tensorcnt)
    __builtin_amdgcn_s_wait_tensorcnt(0);
#else
    asm volatile("s_wait_tensorcnt 0x0" ::: "memory");
#endif
}
#elif defined(HAVE_ASYNC)
static __device__ __forceinline__ void async_b128(
    unsigned short* lds_dst, const unsigned short* gsrc)
{
    __builtin_amdgcn_global_load_async_to_lds_b128(
        (const __attribute__((address_space(1))) void*)gsrc,
        (__attribute__((address_space(3))) void*)lds_dst, 0, 0);
}
static __device__ __forceinline__ void wait_tile_loads() {
#if __has_builtin(__builtin_amdgcn_s_wait_asynccnt)
    __builtin_amdgcn_s_wait_asynccnt(0);
#else
    asm volatile("s_wait_asynccnt 0x0" ::: "memory");
#endif
}
#else
static __device__ __forceinline__ void wait_tile_loads() {}
#endif

// ---------------------------------------------------------------------------
// Zero the histogram (ws is poisoned between runs)
// ---------------------------------------------------------------------------
__global__ void vq_zero_hist(int* __restrict__ hist) {
    hist[blockIdx.x * 256 + threadIdx.x] = 0;
}

// ---------------------------------------------------------------------------
// Row L2-normalize: one wave32 per row of 256 f32.
// ---------------------------------------------------------------------------
__global__ __launch_bounds__(256) void vq_l2norm(
    const float* __restrict__ src,
    unsigned short* __restrict__ dst16,
    float* __restrict__ dst32,          // may be null
    float* __restrict__ norms)          // may be null
{
    const int wave = threadIdx.x >> 5;
    const int lane = threadIdx.x & 31;
    const long row = (long)blockIdx.x * 8 + wave;

    const float* r = src + row * DDIM;
    float v[8];
    float s = 0.f;
#pragma unroll
    for (int j = 0; j < 8; ++j) { v[j] = r[lane + 32 * j]; s += v[j] * v[j]; }
#pragma unroll
    for (int off = 16; off > 0; off >>= 1) s += __shfl_xor(s, off);

    const float nrm = sqrtf(s);
    const float inv = 1.0f / fmaxf(nrm, 1e-12f);
#pragma unroll
    for (int j = 0; j < 8; ++j) {
        const int c = lane + 32 * j;
        const float fn = v[j] * inv;
        dst16[row * DDIM + c] = f32_to_bf16_rne(fn);
        if (dst32) dst32[row * DDIM + c] = fn;
    }
    if (norms && lane == 0) norms[row] = nrm;
}

// ---------------------------------------------------------------------------
// Deterministic sum of n floats -> out[0] = sum * scale   (single block)
// ---------------------------------------------------------------------------
__global__ __launch_bounds__(1024) void vq_reduce_sum(
    const float* __restrict__ src, int n, float scale, float* __restrict__ out)
{
    __shared__ float sm[1024];
    float s = 0.f;
    for (int i = threadIdx.x; i < n; i += 1024) s += src[i];
    sm[threadIdx.x] = s;
    __syncthreads();
    for (int st = 512; st > 0; st >>= 1) {
        if ((int)threadIdx.x < st) sm[threadIdx.x] += sm[threadIdx.x + st];
        __syncthreads();
    }
    if (threadIdx.x == 0) out[0] = sm[0] * scale;
}

// ---------------------------------------------------------------------------
// Fused bf16 WMMA GEMM + running argmax.
// Block = 256 threads (8 waves), 128 rows/block; wave w -> rows [w*16, w*16+16).
// A fragment (K=256) is held in registers (64 VGPRs/lane) for the whole scan.
// B tiles (32 codes x 256 K = 16 KB) stream via the Tensor Data Mover into a
// double buffer; per iteration all 16 B-fragments are preloaded to registers,
// then 16 WMMAs run as two interleaved accumulator chains.
// ---------------------------------------------------------------------------
__global__ __launch_bounds__(256) void vq_gemm_argmax(
    const unsigned short* __restrict__ a16,   // [N][256] bf16 normalized inputs
    const unsigned short* __restrict__ b16,   // [E][256] bf16 normalized codebook
    int* __restrict__ idx_out,                // [N]
    int* __restrict__ hist)                   // [E]
{
    // 64 KB: stages the 128x256 A block, then reused as B double buffer (2 x 16 KB)
    __shared__ alignas(16) unsigned short Smem[128 * DDIM];

    const int tid  = threadIdx.x;
    const int wave = tid >> 5;
    const int lane = tid & 31;
    const int nn   = lane & 15;     // col within 16-tile / M within 16-group
    const int half = lane >> 4;
    const long gr  = (long)blockIdx.x * 128;

    // ---- Stage A block into LDS ----
#if defined(HAVE_TDM)
    if (wave == 0) tdm_load_rows(lds_byte_off(Smem), a16 + gr * DDIM, 128);
    wait_tile_loads();
#elif defined(HAVE_ASYNC)
    {
        const unsigned short* src = a16 + gr * DDIM;
#pragma unroll
        for (int i = 0; i < 32; ++i)
            async_b128(&Smem[(tid + i * 256) * 8], &src[(tid + i * 256) * 8]);
    }
    wait_tile_loads();
#else
    {
        const uint4* src = (const uint4*)(a16 + gr * DDIM);
        uint4* dst = (uint4*)Smem;
#pragma unroll
        for (int i = 0; i < 32; ++i) dst[tid + i * 256] = src[tid + i * 256];
    }
#endif
    __syncthreads();

    // ---- Hoist this wave's A fragment (full K=256) into registers ----
    // A 16x32 layout (contiguous k): lanes 0-15 take K 0..15, lanes 16-31 K 16..31
    const int aoff = (wave * 16 + nn) * DDIM + half * 16;
    uint4 areg[16];
#pragma unroll
    for (int kc = 0; kc < 8; ++kc) {
        areg[2 * kc]     = *(const uint4*)&Smem[aoff + kc * 32];
        areg[2 * kc + 1] = *(const uint4*)&Smem[aoff + kc * 32 + 8];
    }
    __syncthreads();   // A staging free; reuse Smem[0..16383] as B double buffer

    float rmax[8];
    int   ridx[8];
#pragma unroll
    for (int r = 0; r < 8; ++r) { rmax[r] = -3.402823e38f; ridx[r] = 0; }

    const unsigned lds_base = lds_byte_off(Smem);
    const int boff0 = nn * DDIM + half * 16;           // sub-tile 0: codes [0,16)
    const int boff1 = (16 + nn) * DDIM + half * 16;    // sub-tile 1: codes [16,32)

    // Prologue: load B tile 0 (32 codes, 16 KB) into buffer 0
#if defined(HAVE_TDM)
    if (wave == 0) tdm_load_rows(lds_base, b16, 32);
#elif defined(HAVE_ASYNC)
#pragma unroll
    for (int i = 0; i < 4; ++i)
        async_b128(&Smem[(tid + i * 256) * 8], &b16[(tid + i * 256) * 8]);
#else
    {
        const uint4* src = (const uint4*)b16;
        uint4* dst = (uint4*)Smem;
#pragma unroll
        for (int i = 0; i < 4; ++i) dst[tid + i * 256] = src[tid + i * 256];
    }
#endif
    wait_tile_loads();
    __syncthreads();

    for (int ct = 0; ct < 256; ++ct) {     // 32 codes per iteration
        const int cur = ct & 1;

        // Kick off the next 32-code tile into the other buffer (overlaps compute)
        if (ct + 1 < 256) {
            const unsigned short* gsrc = b16 + (size_t)(ct + 1) * (32 * DDIM);
            unsigned short* ldst = Smem + (cur ^ 1) * 8192;
            (void)ldst;
#if defined(HAVE_TDM)
            if (wave == 0) tdm_load_rows(lds_base + (unsigned)(cur ^ 1) * 16384u, gsrc, 32);
#elif defined(HAVE_ASYNC)
#pragma unroll
            for (int i = 0; i < 4; ++i)
                async_b128(&ldst[(tid + i * 256) * 8], &gsrc[(tid + i * 256) * 8]);
#else
#pragma unroll
            for (int i = 0; i < 4; ++i)
                ((uint4*)ldst)[tid + i * 256] = ((const uint4*)gsrc)[tid + i * 256];
#endif
        }

        const unsigned short* Bb = Smem + cur * 8192;

        // Preload all B fragments for both 16-code sub-tiles -> loads pipeline
        uint4 br0[16], br1[16];
#pragma unroll
        for (int kc = 0; kc < 8; ++kc) {
            br0[2 * kc]     = *(const uint4*)&Bb[boff0 + kc * 32];
            br0[2 * kc + 1] = *(const uint4*)&Bb[boff0 + kc * 32 + 8];
            br1[2 * kc]     = *(const uint4*)&Bb[boff1 + kc * 32];
            br1[2 * kc + 1] = *(const uint4*)&Bb[boff1 + kc * 32 + 8];
        }

        v8f acc0 = {0.f, 0.f, 0.f, 0.f, 0.f, 0.f, 0.f, 0.f};
        v8f acc1 = {0.f, 0.f, 0.f, 0.f, 0.f, 0.f, 0.f, 0.f};
#pragma unroll
        for (int kc = 0; kc < 8; ++kc) {     // two independent WMMA chains
            U32x8 au{areg[2 * kc], areg[2 * kc + 1]};
            U32x8 bu0{br0[2 * kc], br0[2 * kc + 1]};
            U32x8 bu1{br1[2 * kc], br1[2 * kc + 1]};
            const v16bf Av  = __builtin_bit_cast(v16bf, au);
            const v16bf Bv0 = __builtin_bit_cast(v16bf, bu0);
            const v16bf Bv1 = __builtin_bit_cast(v16bf, bu1);
            acc0 = __builtin_amdgcn_wmma_f32_16x16x32_bf16(
                false, Av, false, Bv0, (short)0, acc0, false, false);
            acc1 = __builtin_amdgcn_wmma_f32_16x16x32_bf16(
                false, Av, false, Bv1, (short)0, acc1, false, false);
        }

        const int col0 = ct * 32 + nn;       // ascending order: first tie wins
        const int col1 = col0 + 16;
#pragma unroll
        for (int r = 0; r < 8; ++r) {
            if (acc0[r] > rmax[r]) { rmax[r] = acc0[r]; ridx[r] = col0; }
        }
#pragma unroll
        for (int r = 0; r < 8; ++r) {
            if (acc1[r] > rmax[r]) { rmax[r] = acc1[r]; ridx[r] = col1; }
        }

        wait_tile_loads();   // next tile resident (TENSORcnt/ASYNCcnt -> 0)
        __syncthreads();     // all waves done with `cur` before it is overwritten
    }

    // Cross-lane argmax within each 16-lane half (xor 1,2,4,8 stays in half).
#pragma unroll
    for (int r = 0; r < 8; ++r) {
        float m_ = rmax[r];
        int   i_ = ridx[r];
        for (int off = 1; off < 16; off <<= 1) {
            const float om = __shfl_xor(m_, off);
            const int   oi = __shfl_xor(i_, off);
            if (om > m_ || (om == m_ && oi < i_)) { m_ = om; i_ = oi; }  // smallest idx on ties
        }
        if (nn == 0) {
            const long row = gr + wave * 16 + half * 8 + r;
            idx_out[row] = i_;
            atomicAdd(&hist[i_], 1);   // integer atomic: order-independent, deterministic
        }
    }
}

// ---------------------------------------------------------------------------
// Gather + scale + straight-through output + per-row squared-error partials.
// ---------------------------------------------------------------------------
__global__ __launch_bounds__(256) void vq_quantize(
    const float* __restrict__ x,
    const float* __restrict__ codef32,
    const int*   __restrict__ idx,
    const float* __restrict__ scal,      // scal[0] = mean input scale
    float* __restrict__ out,
    float* __restrict__ sqpart)
{
    const int wave = threadIdx.x >> 5;
    const int lane = threadIdx.x & 31;
    const long row = (long)blockIdx.x * 8 + wave;

    const float ms = scal[0];
    const int e = idx[row];
    const float* xr = x + row * DDIM;
    const float* cr = codef32 + (size_t)e * DDIM;

    float v[8];
    float s = 0.f;
#pragma unroll
    for (int j = 0; j < 8; ++j) { v[j] = xr[lane + 32 * j]; s += v[j] * v[j]; }
#pragma unroll
    for (int off = 16; off > 0; off >>= 1) s += __shfl_xor(s, off);

    const float inv = 1.0f / fmaxf(sqrtf(s), 1e-12f);
    float sq = 0.f;
#pragma unroll
    for (int j = 0; j < 8; ++j) {
        const int c = lane + 32 * j;
        const float fn = v[j] * inv;
        const float qn = cr[c];
        const float q  = qn * ms;
        out[row * DDIM + c] = v[j] + (q - v[j]);   // match straight-through float rounding
        const float d = qn - fn;
        sq += d * d;
    }
#pragma unroll
    for (int off = 16; off > 0; off >>= 1) sq += __shfl_xor(sq, off);
    if (lane == 0) sqpart[row] = sq;
}

// ---------------------------------------------------------------------------
// Perplexity from histogram (single block, deterministic)
// ---------------------------------------------------------------------------
__global__ __launch_bounds__(256) void vq_perplexity(
    const int* __restrict__ hist, float* __restrict__ out)
{
    __shared__ float sm[256];
    float s = 0.f;
    for (int e = threadIdx.x; e < ECODES; e += 256) {
        const float p = (float)hist[e] * (1.0f / (float)NROWS);
        s += p * logf(p + 1e-10f);
    }
    sm[threadIdx.x] = s;
    __syncthreads();
    for (int st = 128; st > 0; st >>= 1) {
        if ((int)threadIdx.x < st) sm[threadIdx.x] += sm[threadIdx.x + st];
        __syncthreads();
    }
    if (threadIdx.x == 0) out[0] = expf(-sm[0]);
}

// ---------------------------------------------------------------------------
extern "C" void kernel_launch(void* const* d_in, const int* in_sizes, int n_in,
                              void* d_out, int out_size, void* d_ws, size_t ws_size,
                              hipStream_t stream)
{
    (void)in_sizes; (void)n_in; (void)out_size; (void)ws_size;

    const float* x  = (const float*)d_in[0];   // [32,1024,256] f32
    const float* cb = (const float*)d_in[1];   // [8192,256] f32
    float* out = (float*)d_out;                // [N*D] quantized, then loss, then perplexity

    char* ws = (char*)d_ws;
    unsigned short* flatn16 = (unsigned short*)(ws + OFF_FLATN16);
    unsigned short* code16  = (unsigned short*)(ws + OFF_CODE16);
    float* code32 = (float*)(ws + OFF_CODE32);
    int*   idxbuf = (int*)  (ws + OFF_IDX);
    int*   hist   = (int*)  (ws + OFF_HIST);
    float* norms  = (float*)(ws + OFF_NORMS);
    float* sqpart = (float*)(ws + OFF_SQ);
    float* scal   = (float*)(ws + OFF_SCAL);

    vq_zero_hist<<<dim3(ECODES / 256), dim3(256), 0, stream>>>(hist);
    vq_l2norm<<<dim3(NROWS / 8), dim3(256), 0, stream>>>(x, flatn16, nullptr, norms);
    vq_l2norm<<<dim3(ECODES / 8), dim3(256), 0, stream>>>(cb, code16, code32, nullptr);
    vq_reduce_sum<<<dim3(1), dim3(1024), 0, stream>>>(norms, NROWS, 1.0f / (float)NROWS, scal);
    vq_gemm_argmax<<<dim3(NROWS / 128), dim3(256), 0, stream>>>(flatn16, code16, idxbuf, hist);
    vq_quantize<<<dim3(NROWS / 8), dim3(256), 0, stream>>>(x, code32, idxbuf, scal, out, sqpart);
    vq_reduce_sum<<<dim3(1), dim3(1024), 0, stream>>>(
        sqpart, NROWS, 1.25f / ((float)NROWS * (float)DDIM), out + NDOUT);
    vq_perplexity<<<dim3(1), dim3(256), 0, stream>>>(hist, out + NDOUT + 1);
}
